// VectorQuantizer_60670708023852
// MI455X (gfx1250) — matrix operand order repaired and checked by reading, not compiled, not run on hardware
//
#include <hip/hip_runtime.h>

// ---------------------------------------------------------------------------
// VQ-VAE vector quantizer for MI455X (gfx1250).
// Dominant cost: 32768x8192x256 distance GEMM (137 GFLOP) -> bf16 WMMA with
// fused per-tile argmin. Epilogue kernels are bandwidth-bound (~90MB ~ 4us).
// Chunk staging uses the Tensor Data Mover (tensor_load_to_lds, TENSORcnt)
// double-buffered in LDS so DMA overlaps WMMA compute.
// ---------------------------------------------------------------------------

#define DIM   256
#define KCB   8192
#define NTOK  32768

#define EMA_DECAY 0.99f
#define ONE_M_D   0.01f
#define EPS_F     1e-5f
#define USAGE_THRESH 4.3171247411657864e-05f   // 0.99^1000

typedef __bf16 bf16_t;
typedef bf16_t v16bf __attribute__((ext_vector_type(16)));
typedef bf16_t v8bf  __attribute__((ext_vector_type(8)));
typedef float  v8f   __attribute__((ext_vector_type(8)));
typedef unsigned int v4u __attribute__((ext_vector_type(4)));
typedef int v8i __attribute__((ext_vector_type(8)));
typedef int v4i __attribute__((ext_vector_type(4)));

#if defined(__has_builtin)
#  if __has_builtin(__builtin_amdgcn_tensor_load_to_lds) && \
      __has_builtin(__builtin_amdgcn_s_wait_tensorcnt)
#    define USE_TDM 1
#  endif
#endif
#ifndef USE_TDM
#  define USE_TDM 0
#endif

// ---------------- block-wide sum reduction (256 threads, wave32) -----------
__device__ __forceinline__ float blockReduceSum256(float v) {
    #pragma unroll
    for (int off = 16; off >= 1; off >>= 1) v += __shfl_down(v, off, 32);
    __shared__ float red[8];
    const int wave = threadIdx.x >> 5, lane = threadIdx.x & 31;
    if (lane == 0) red[wave] = v;
    __syncthreads();
    if (wave == 0) {
        v = (lane < 8) ? red[lane] : 0.0f;
        #pragma unroll
        for (int off = 4; off >= 1; off >>= 1) v += __shfl_down(v, off, 32);
    }
    return v;   // valid in thread 0
}

// ---------------- prep: embedding -> bf16 copy + fp32 row norms ------------
__global__ __launch_bounds__(256) void vq_prep_emb(const float* __restrict__ emb,
                                                   bf16_t* __restrict__ eb,
                                                   float* __restrict__ en) {
    const int k = blockIdx.x, d = threadIdx.x;
    const float v = emb[(size_t)k * DIM + d];
    eb[(size_t)k * DIM + d] = (bf16_t)v;
    const float s = blockReduceSum256(v * v);
    if (threadIdx.x == 0) en[k] = s;
}

// ---------------- zero the accumulation regions (graph-replay safe) --------
__global__ __launch_bounds__(256) void vq_init(float* __restrict__ w_acc,
                                               float* __restrict__ cs_acc,
                                               float* __restrict__ lossacc,
                                               float* __restrict__ ntot) {
    const size_t i = (size_t)blockIdx.x * 256 + threadIdx.x;
    const size_t stride = (size_t)gridDim.x * 256;
    for (size_t p = i; p < (size_t)KCB * DIM; p += stride) w_acc[p] = 0.0f;
    if (i < KCB) cs_acc[i] = 0.0f;
    if (i == 0) { *lossacc = 0.0f; *ntot = 0.0f; }
}

// ---------------- main: fused distance GEMM + argmin -----------------------
// Block: 256 thr = 8 waves, 128 rows of x per block (16 per wave).
// K is processed in chunks of 64 codebook rows staged to LDS (bf16).
#define CHUNK    64
#define PITCH_B  544                   // bytes/LDS row: 512 data + 32 pad
#define BUFBYTES (CHUNK * PITCH_B)     // 34816 B per buffer

#if USE_TDM
// Tensor DMA: 2-D tile, data_size=8B. tensor/tile dim0 = 64 (512B row),
// 64 rows, contiguous stride. Pad: after every 128 DWORDs (=512B row) insert
// 8 DWORDs (32B) into the LDS address stream -> matches PITCH_B = 544.
__device__ __forceinline__ void tdm_load_chunk(const bf16_t* gsrc, unsigned lds_off) {
    const unsigned long long ga = (unsigned long long)(uintptr_t)gsrc;
    v4u g0;
    g0[0] = 1u;                                            // count=1, user D#
    g0[1] = lds_off;                                       // lds_addr
    g0[2] = (unsigned)(ga & 0xFFFFFFFFull);                // global_addr[31:0]
    g0[3] = (unsigned)((ga >> 32) & 0x1FFFFFFull)          // global_addr[56:32]
          | (2u << 30);                                    // type = 2 (image)
    const unsigned flags = (3u << 16)                      // data_size = 8B
                         | (1u << 20)                      // pad_enable
                         | (6u << 22)                      // pad_interval: 128 dw
                         | (7u << 25);                     // pad_amount: 8 dw
    const unsigned td0 = 64u, td1 = 64u;                   // tensor dims (8B units)
    const unsigned tile0 = 64u, tile1 = 64u;               // tile dims
    const unsigned str0 = 64u;                             // dim0 stride (8B units)
    v8i g1;
    g1[0] = (int)flags;                                    // mask=0 | flags
    g1[1] = (int)((td0 & 0xFFFFu) << 16);                  // abar=0 | td0.lo
    g1[2] = (int)(((td0 >> 16) & 0xFFFFu) | ((td1 & 0xFFFFu) << 16));
    g1[3] = (int)(((td1 >> 16) & 0xFFFFu) | (tile0 << 16));
    g1[4] = (int)(tile1);                                  // tile1 | tile2=0
    g1[5] = (int)str0;                                     // stride0[31:0]
    g1[6] = 0;                                             // stride0.hi|stride1.lo
    g1[7] = 0;                                             // stride1.hi
    const v4i gz4 = {0, 0, 0, 0};                          // 2-D: groups 2/3 unused
    const v8i gz8 = {0, 0, 0, 0, 0, 0, 0, 0};              // VADDR4: unused/NULL
    // 6-arg toolchain flavor: (g0, g1, g2, g3, g4, cpol)
    __builtin_amdgcn_tensor_load_to_lds(g0, g1, gz4, gz4, gz8, 0);
}
#endif

__global__ __launch_bounds__(256) void vq_argmin(const float* __restrict__ x,
                                                 const bf16_t* __restrict__ eb,
                                                 const float* __restrict__ en,
                                                 float* __restrict__ out_idx_f,
                                                 int* __restrict__ widx) {
#if USE_TDM
    __shared__ __align__(16) unsigned char sB[2 * BUFBYTES];   // double buffer
#else
    __shared__ __align__(16) unsigned char sB[BUFBYTES];
#endif

    const int tid  = threadIdx.x;
    const int wave = tid >> 5;
    const int lane = tid & 31;
    const int l15  = lane & 15;
    const int half = lane >> 4;
    const int rowBase = blockIdx.x * 128 + wave * 16;

    // --- preload A fragments: this wave's 16 x-rows, full DIM (8 k-steps) ---
    // A 16x32 layout (matches B): lane l<16 holds row l, K = k0..k0+15 contiguous
    // with k0 = half*16 inside each 32-wide step.
    v16bf a[8];
    const float* xrow = x + (size_t)(rowBase + l15) * DIM;
    #pragma unroll
    for (int ks = 0; ks < 8; ++ks) {
        const int k0 = ks * 32 + half * 16;
        const float* p0 = xrow + k0;
        const float* p1 = xrow + k0 + 8;
        v16bf t;
        #pragma unroll
        for (int i = 0; i < 8; ++i) {
            t[i]     = (bf16_t)p0[i];
            t[i + 8] = (bf16_t)p1[i];
        }
        a[ks] = t;
    }

    float best[8];
    int   bidx[8];
    #pragma unroll
    for (int j = 0; j < 8; ++j) { best[j] = 3.4028235e38f; bidx[j] = 0; }

    const int NCHUNK = KCB / CHUNK;

#if USE_TDM
    const bool dmaWave = (wave == 0);
    const unsigned ldsBase = (unsigned)(uintptr_t)(void*)sB;   // flat addr[31:0] = LDS offset
    if (dmaWave) {
        tdm_load_chunk(eb, ldsBase);                           // chunk 0 -> buf 0
        __builtin_amdgcn_s_wait_tensorcnt(0);
    }
    __syncthreads();
#endif

    for (int c = 0; c < NCHUNK; ++c) {
        const int n0 = c * CHUNK;
#if USE_TDM
        const unsigned char* buf = sB + (unsigned)(c & 1) * BUFBYTES;
        // overlap: DMA next chunk into the other buffer while we compute
        if (dmaWave && (c + 1) < NCHUNK)
            tdm_load_chunk(eb + (size_t)(n0 + CHUNK) * DIM,
                           ldsBase + (unsigned)((c + 1) & 1) * BUFBYTES);
#else
        const unsigned char* buf = sB;
        if (c + 1 < NCHUNK)
            __builtin_prefetch((const char*)(eb + (size_t)(n0 + CHUNK) * DIM) + tid * 128, 0, 1);
        {   // cooperative staging: 64 rows x 512B -> padded LDS
            const uint4* src = (const uint4*)(eb + (size_t)n0 * DIM);  // 2048 uint4
            #pragma unroll
            for (int i = 0; i < 8; ++i) {
                const int q = tid + i * 256;
                const int r = q >> 5, cc = q & 31;
                uint4 v = src[q];
                *(uint4*)(sB + r * PITCH_B + cc * 16) = v;
            }
        }
        __syncthreads();
#endif

        // --- 4 N-tiles per wave, 8 k-steps; B frags double-buffered in regs ---
        // B 32x16 layout used here (contiguous k, same as A): lane l<16 holds column l, K = k0..k0+15
        // contiguous; lane l>=16 holds column l-16, K = k0+16..k0+31.
        v8f acc[4];
        #pragma unroll
        for (int nt = 0; nt < 4; ++nt) acc[nt] = (v8f){0, 0, 0, 0, 0, 0, 0, 0};

        const unsigned char* brow = buf + l15 * PITCH_B + half * 32;
        v16bf bc[4];
        #pragma unroll
        for (int nt = 0; nt < 4; ++nt)
            bc[nt] = *(const v16bf*)(brow + nt * 16 * PITCH_B);

        #pragma unroll
        for (int ks = 0; ks < 8; ++ks) {
            v16bf bn[4];
            if (ks < 7) {
                #pragma unroll
                for (int nt = 0; nt < 4; ++nt)
                    bn[nt] = *(const v16bf*)(brow + nt * 16 * PITCH_B + (ks + 1) * 64);
            }
            #pragma unroll
            for (int nt = 0; nt < 4; ++nt)
                acc[nt] = __builtin_amdgcn_wmma_f32_16x16x32_bf16(
                    false, a[ks], false, bc[nt], (short)0, acc[nt], false, false);
            #pragma unroll
            for (int nt = 0; nt < 4; ++nt) bc[nt] = bn[nt];
        }

        // --- fused argmin epilogue: dist = ||e||^2 - 2 x.e (||x||^2 const) ---
        #pragma unroll
        for (int nt = 0; nt < 4; ++nt) {
            const int n = n0 + nt * 16 + l15;
            const float env = en[n];
            #pragma unroll
            for (int j = 0; j < 8; ++j) {
                const float d = env - 2.0f * acc[nt][j];
                if (d < best[j]) { best[j] = d; bidx[j] = n; }
            }
        }

#if USE_TDM
        if (dmaWave) __builtin_amdgcn_s_wait_tensorcnt(0);
        __syncthreads();
#else
        __syncthreads();
#endif
    }

    // --- cross-lane argmin within each 16-lane half (C/D layout: rows j / j+8)
    #pragma unroll
    for (int j = 0; j < 8; ++j) {
        #pragma unroll
        for (int off = 8; off >= 1; off >>= 1) {
            const float ov = __shfl_xor(best[j], off, 32);
            const int   oi = __shfl_xor(bidx[j], off, 32);
            if (ov < best[j] || (ov == best[j] && oi < bidx[j])) {
                best[j] = ov; bidx[j] = oi;
            }
        }
    }
    if (l15 == 0) {
        #pragma unroll
        for (int j = 0; j < 8; ++j) {
            const int m = rowBase + half * 8 + j;
            widx[m] = bidx[j];
            out_idx_f[m] = (float)bidx[j];
        }
    }
}

// ---------------- gather + commit loss + EMA scatter-adds ------------------
__global__ __launch_bounds__(256) void vq_quant_scatter(const float* __restrict__ x,
                                                        const float* __restrict__ emb,
                                                        const float* __restrict__ scale,
                                                        const int* __restrict__ widx,
                                                        float* __restrict__ out_q,
                                                        float* __restrict__ w_acc,
                                                        float* __restrict__ cs_acc,
                                                        float* __restrict__ lossacc) {
    const int t = blockIdx.x, d = threadIdx.x;
    const int idx = widx[t];
    const float s  = scale[0];
    const float e  = emb[(size_t)idx * DIM + d] * s;    // quantized value
    const float xv = x[(size_t)t * DIM + d];
    out_q[(size_t)t * DIM + d] = e;                     // STE value == quantized
    const float diff = e - xv;
    atomicAdd(&w_acc[(size_t)idx * DIM + d], ONE_M_D * xv);   // (1-decay)*dw
    const float part = blockReduceSum256(diff * diff);
    if (threadIdx.x == 0) {
        atomicAdd(lossacc, part);
        atomicAdd(&cs_acc[idx], ONE_M_D);               // (1-decay)*n_i
    }
}

// ---------------- EMA cluster-size combine + n_tot reduction ---------------
__global__ __launch_bounds__(256) void vq_cs_combine(const float* __restrict__ ema_cs,
                                                     float* __restrict__ cs_out,
                                                     float* __restrict__ ntot) {
    const int k = blockIdx.x * 256 + threadIdx.x;
    const float v = EMA_DECAY * ema_cs[k] + cs_out[k];
    cs_out[k] = v;
    const float s = blockReduceSum256(v);
    if (threadIdx.x == 0) atomicAdd(ntot, s);
}

// ---------------- finalize: smoothing, dead-code restart, loss -------------
__global__ __launch_bounds__(256) void vq_finalize(const float* __restrict__ x,
                                                   const float* __restrict__ ema_w,
                                                   const float* __restrict__ ntot,
                                                   const float* __restrict__ lossacc,
                                                   float* __restrict__ emb_out,
                                                   float* __restrict__ w_out,
                                                   float* __restrict__ cs_out,
                                                   float* __restrict__ loss_out) {
    const int k = blockIdx.x, d = threadIdx.x;
    const float cs = cs_out[k];
    const float nt = *ntot;
    const float nw = EMA_DECAY * ema_w[(size_t)k * DIM + d] + w_out[(size_t)k * DIM + d];
    const float cssm = (cs + EPS_F) / (nt + (float)KCB * EPS_F) * nt;
    const bool dead = cs < USAGE_THRESH;
    // deterministic stand-in for jax randint(key(1)) replacement rows
    const unsigned r = (1103515245u * (unsigned)k + 12345u) & (NTOK - 1);
    const float repl = x[(size_t)r * DIM + d];
    emb_out[(size_t)k * DIM + d] = dead ? repl : nw / cssm;
    w_out[(size_t)k * DIM + d]   = dead ? repl : nw;
    if (d == 0) cs_out[k] = dead ? 1.0f : cs;
    if (k == 0 && d == 0) *loss_out = *lossacc * (1.0f / ((float)NTOK * (float)DIM));
}

// ---------------------------------------------------------------------------
extern "C" void kernel_launch(void* const* d_in, const int* in_sizes, int n_in,
                              void* d_out, int out_size, void* d_ws, size_t ws_size,
                              hipStream_t stream) {
    const float* x       = (const float*)d_in[0];
    const float* scale   = (const float*)d_in[1];
    const float* emb     = (const float*)d_in[2];
    const float* ema_cs  = (const float*)d_in[3];
    const float* ema_w   = (const float*)d_in[4];

    // output tuple, concatenated flat:
    float* out      = (float*)d_out;
    float* out_q    = out;                                  // N*DIM
    float* out_idx  = out_q + (size_t)NTOK * DIM;           // N
    float* out_loss = out_idx + NTOK;                       // 1
    float* out_emb  = out_loss + 1;                         // K*DIM
    float* out_cs   = out_emb + (size_t)KCB * DIM;          // K
    float* out_w    = out_cs + KCB;                         // K*DIM

    // workspace layout (~4.4 MB)
    char*   ws      = (char*)d_ws;
    bf16_t* eb      = (bf16_t*)ws;                          // K*DIM bf16 = 4 MB
    float*  en      = (float*)(ws + (size_t)KCB * DIM * 2); // K
    int*    widx    = (int*)((char*)en + KCB * 4);          // N
    float*  lossacc = (float*)((char*)widx + NTOK * 4);
    float*  ntotp   = lossacc + 1;

    vq_prep_emb   <<<KCB,        256, 0, stream>>>(emb, eb, en);
    vq_init       <<<2048,       256, 0, stream>>>(out_w, out_cs, lossacc, ntotp);
    vq_argmin     <<<NTOK / 128, 256, 0, stream>>>(x, eb, en, out_idx, widx);
    vq_quant_scatter<<<NTOK,     256, 0, stream>>>(x, emb, scale, widx,
                                                   out_q, out_w, out_cs, lossacc);
    vq_cs_combine <<<KCB / 256,  256, 0, stream>>>(ema_cs, out_cs, ntotp);
    vq_finalize   <<<KCB,        256, 0, stream>>>(x, ema_w, ntotp, lossacc,
                                                   out_emb, out_w, out_cs, out_loss);
}
